// MoeFFN_7636451852397
// MI455X (gfx1250) — matrix operand order repaired and checked by reading, not compiled, not run on hardware
//
#include <hip/hip_runtime.h>
#include <math.h>

#define N_TOK   4096
#define DMODEL  1024
#define IFFN    4096
#define NEXP    8
#define TILE_M  32
#define TILES_PER_E 128   // N_TOK / TILE_M

typedef __attribute__((ext_vector_type(16))) __bf16 v16bf;
typedef __attribute__((ext_vector_type(8)))  float  v8f;

union FragAB { v16bf v; uint4 u[2]; };

__device__ __forceinline__ unsigned short f32_bf16(float f) {
  unsigned int u = __float_as_uint(f);
  u += 0x7FFFu + ((u >> 16) & 1u);            // round-to-nearest-even
  return (unsigned short)(u >> 16);
}

__device__ __forceinline__ float gelu_exact(float v) {
  return 0.5f * v * (1.0f + erff(v * 0.70710678118f));
}

// ---------------------------------------------------------------- utilities
__global__ void moe_zero_counts(int* counts) {
  if (threadIdx.x < NEXP) counts[threadIdx.x] = 0;
}

__global__ void moe_cvt_bf16(const float4* __restrict__ s, uint2* __restrict__ d, int n4) {
  int i = blockIdx.x * blockDim.x + threadIdx.x;
  int stride = gridDim.x * blockDim.x;
  for (; i < n4; i += stride) {
    float4 f = s[i];
    uint2 o;
    o.x = (unsigned)f32_bf16(f.x) | ((unsigned)f32_bf16(f.y) << 16);
    o.y = (unsigned)f32_bf16(f.z) | ((unsigned)f32_bf16(f.w) << 16);
    d[i] = o;
  }
}

// ---------------------------------------------------------------- gating + routing
__global__ void moe_gate(const float* __restrict__ x, const float* __restrict__ gw,
                         const float* __restrict__ gb, int* __restrict__ counts,
                         int* __restrict__ ltok, float* __restrict__ lwt) {
  int t = blockIdx.x * blockDim.x + threadIdx.x;
  if (t >= N_TOK) return;
  float acc[NEXP];
  #pragma unroll
  for (int e = 0; e < NEXP; ++e) acc[e] = gb[e];
  const float4* xr = (const float4*)(x + (size_t)t * DMODEL);
  for (int j = 0; j < DMODEL / 4; ++j) {
    float4 xv = xr[j];
    #pragma unroll
    for (int e = 0; e < NEXP; ++e) {
      float4 g = ((const float4*)(gw + e * DMODEL))[j];
      acc[e] += xv.x * g.x + xv.y * g.y + xv.z * g.z + xv.w * g.w;
    }
  }
  int i1 = 0; float v1 = acc[0];
  #pragma unroll
  for (int e = 1; e < NEXP; ++e) if (acc[e] > v1) { v1 = acc[e]; i1 = e; }
  int i2 = (i1 == 0) ? 1 : 0; float v2 = -3.4e38f;
  #pragma unroll
  for (int e = 0; e < NEXP; ++e) if (e != i1 && acc[e] > v2) { v2 = acc[e]; i2 = e; }
  float ex = __expf(v2 - v1);
  float inv = 1.0f / (1.0f + ex);
  int p1 = atomicAdd(&counts[i1], 1);
  ltok[i1 * N_TOK + p1] = (t << 1);
  lwt [i1 * N_TOK + p1] = inv;
  int p2 = atomicAdd(&counts[i2], 1);
  ltok[i2 * N_TOK + p2] = (t << 1) | 1;
  lwt [i2 * N_TOK + p2] = ex * inv;
}

// ---------------------------------------------------------------- fused expert FFN (WMMA bf16)
// One block = 32 tokens of one expert (two 16-row M-tiles sharing B fragments).
__global__ void __launch_bounds__(256)
moe_ffn_wmma(const unsigned short* __restrict__ w1bf,
             const unsigned short* __restrict__ w2bf,
             const unsigned short* __restrict__ xbf,
             const float* __restrict__ b1, const float* __restrict__ b2,
             const int* __restrict__ counts,
             const int* __restrict__ ltok, const float* __restrict__ lwt,
             float* __restrict__ slots) {
  extern __shared__ unsigned short hlds[];      // [TILE_M][IFFN] bf16 = 256 KB
  __shared__ int   s_pair [TILE_M];
  __shared__ int   s_tok  [TILE_M];
  __shared__ int   s_valid[TILE_M];
  __shared__ float s_wt   [TILE_M];

  const int e    = blockIdx.x / TILES_PER_E;
  const int tile = blockIdx.x % TILES_PER_E;
  const int cnt  = counts[e];
  const int base = tile * TILE_M;
  if (base >= cnt) return;                      // empty tile slot -> cheap exit

  if (threadIdx.x < TILE_M) {
    int idx    = base + (int)threadIdx.x;
    int valid  = idx < cnt;
    int packed = valid ? ltok[e * N_TOK + idx] : 0;
    s_pair [threadIdx.x] = packed;              // token*2 + k  -> slot id
    s_tok  [threadIdx.x] = packed >> 1;
    s_wt   [threadIdx.x] = valid ? lwt[e * N_TOK + idx] : 0.0f;
    s_valid[threadIdx.x] = valid;
  }
  __syncthreads();

  const int wave = threadIdx.x >> 5;            // wave32: 8 waves / block
  const int lane = threadIdx.x & 31;
  const int half = lane >> 4;                   // lane-half selects K sub-range
  const int lrow = lane & 15;                   // A: row (token), B: col

  const unsigned short* xrow0 = xbf + (size_t)s_tok[lrow]      * DMODEL;
  const unsigned short* xrow1 = xbf + (size_t)s_tok[16 + lrow] * DMODEL;

  // -------- Phase A: h[32, IFFN] = gelu(x @ w1^T + b1), bf16 into LDS --------
  for (int nt = 0; nt < 32; nt += 2) {
    int col0 = wave * 512 + nt * 16 + lrow;
    int col1 = col0 + 16;
    const unsigned short* wr0 = w1bf + ((size_t)e * IFFN + col0) * DMODEL;
    const unsigned short* wr1 = w1bf + ((size_t)e * IFFN + col1) * DMODEL;
    v8f a00 = {}; v8f a01 = {}; v8f a10 = {}; v8f a11 = {};
    #pragma unroll 4
    for (int kk = 0; kk < DMODEL; kk += 32) {
      FragAB fa0, fa1, fb0, fb1;
      int ka = kk + half * 8;                   // A: K = {0..7} and {16..23} per lane-half
      fa0.u[0] = *(const uint4*)(xrow0 + ka);
      fa0.u[1] = *(const uint4*)(xrow0 + ka + 16);
      fa1.u[0] = *(const uint4*)(xrow1 + ka);
      fa1.u[1] = *(const uint4*)(xrow1 + ka + 16);
      int kb = kk + half * 8;                   // B: K = {0..7} and {16..23} per lane-half
      fb0.u[0] = *(const uint4*)(wr0 + kb);
      fb0.u[1] = *(const uint4*)(wr0 + kb + 16);
      fb1.u[0] = *(const uint4*)(wr1 + kb);
      fb1.u[1] = *(const uint4*)(wr1 + kb + 16);
      a00 = __builtin_amdgcn_wmma_f32_16x16x32_bf16(false, fa0.v, false, fb0.v, (short)0, a00, false, false);
      a01 = __builtin_amdgcn_wmma_f32_16x16x32_bf16(false, fa0.v, false, fb1.v, (short)0, a01, false, false);
      a10 = __builtin_amdgcn_wmma_f32_16x16x32_bf16(false, fa1.v, false, fb0.v, (short)0, a10, false, false);
      a11 = __builtin_amdgcn_wmma_f32_16x16x32_bf16(false, fa1.v, false, fb1.v, (short)0, a11, false, false);
    }
    float bias0 = b1[e * IFFN + col0];
    float bias1 = b1[e * IFFN + col1];
    #pragma unroll
    for (int r = 0; r < 8; ++r) {
      int row = r + half * 8;                   // C/D layout: vgpr r -> M = r (+8 upper lanes)
      hlds[ row       * IFFN + col0] = f32_bf16(gelu_exact(a00[r] + bias0));
      hlds[ row       * IFFN + col1] = f32_bf16(gelu_exact(a01[r] + bias1));
      hlds[(row + 16) * IFFN + col0] = f32_bf16(gelu_exact(a10[r] + bias0));
      hlds[(row + 16) * IFFN + col1] = f32_bf16(gelu_exact(a11[r] + bias1));
    }
  }
  __syncthreads();

  // -------- Phase B: out[32, DMODEL] = (h @ w2^T + b2) * route_w --------
  const unsigned short* hrow0 = hlds + (size_t) lrow       * IFFN;
  const unsigned short* hrow1 = hlds + (size_t)(lrow + 16) * IFFN;
  for (int nt = 0; nt < 8; nt += 2) {
    int d0 = wave * 128 + nt * 16 + lrow;
    int d1 = d0 + 16;
    const unsigned short* wr0 = w2bf + ((size_t)e * DMODEL + d0) * IFFN;
    const unsigned short* wr1 = w2bf + ((size_t)e * DMODEL + d1) * IFFN;
    v8f a00 = {}; v8f a01 = {}; v8f a10 = {}; v8f a11 = {};
    #pragma unroll 4
    for (int kk = 0; kk < IFFN; kk += 32) {
      FragAB fa0, fa1, fb0, fb1;
      int ka = kk + half * 8;
      fa0.u[0] = *(const uint4*)(hrow0 + ka);          // LDS b128 reads
      fa0.u[1] = *(const uint4*)(hrow0 + ka + 16);
      fa1.u[0] = *(const uint4*)(hrow1 + ka);
      fa1.u[1] = *(const uint4*)(hrow1 + ka + 16);
      int kb = kk + half * 8;
      fb0.u[0] = *(const uint4*)(wr0 + kb);
      fb0.u[1] = *(const uint4*)(wr0 + kb + 16);
      fb1.u[0] = *(const uint4*)(wr1 + kb);
      fb1.u[1] = *(const uint4*)(wr1 + kb + 16);
      a00 = __builtin_amdgcn_wmma_f32_16x16x32_bf16(false, fa0.v, false, fb0.v, (short)0, a00, false, false);
      a01 = __builtin_amdgcn_wmma_f32_16x16x32_bf16(false, fa0.v, false, fb1.v, (short)0, a01, false, false);
      a10 = __builtin_amdgcn_wmma_f32_16x16x32_bf16(false, fa1.v, false, fb0.v, (short)0, a10, false, false);
      a11 = __builtin_amdgcn_wmma_f32_16x16x32_bf16(false, fa1.v, false, fb1.v, (short)0, a11, false, false);
    }
    float bias0 = b2[e * DMODEL + d0];
    float bias1 = b2[e * DMODEL + d1];
    #pragma unroll
    for (int r = 0; r < 8; ++r) {
      int row0 = r + half * 8;
      int row1 = row0 + 16;
      if (s_valid[row0]) {
        float w = s_wt[row0];
        slots[(size_t)s_pair[row0] * DMODEL + d0] = (a00[r] + bias0) * w;
        slots[(size_t)s_pair[row0] * DMODEL + d1] = (a01[r] + bias1) * w;
      }
      if (s_valid[row1]) {
        float w = s_wt[row1];
        slots[(size_t)s_pair[row1] * DMODEL + d0] = (a10[r] + bias0) * w;
        slots[(size_t)s_pair[row1] * DMODEL + d1] = (a11[r] + bias1) * w;
      }
    }
  }
}

// ---------------------------------------------------------------- combine two expert slots
__global__ void moe_combine(const float4* __restrict__ slots, float4* __restrict__ out) {
  int i = blockIdx.x * blockDim.x + threadIdx.x;   // N_TOK * (DMODEL/4)
  int t = i >> 8;                                  // DMODEL/4 == 256
  int j = i & 255;
  float4 a = slots[(size_t)(2 * t) * 256 + j];
  float4 b = slots[(size_t)(2 * t + 1) * 256 + j];
  float4 o; o.x = a.x + b.x; o.y = a.y + b.y; o.z = a.z + b.z; o.w = a.w + b.w;
  out[i] = o;
}

// ---------------------------------------------------------------- launcher
extern "C" void kernel_launch(void* const* d_in, const int* in_sizes, int n_in,
                              void* d_out, int out_size, void* d_ws, size_t ws_size,
                              hipStream_t stream) {
  (void)in_sizes; (void)n_in; (void)out_size; (void)ws_size;
  const float* x  = (const float*)d_in[0];
  const float* gw = (const float*)d_in[1];
  const float* gb = (const float*)d_in[2];
  const float* w1 = (const float*)d_in[3];
  const float* b1 = (const float*)d_in[4];
  const float* w2 = (const float*)d_in[5];
  const float* b2 = (const float*)d_in[6];
  float* out = (float*)d_out;

  char* ws = (char*)d_ws;
  size_t off = 0;
  unsigned short* w1bf = (unsigned short*)(ws + off); off += (size_t)NEXP * IFFN * DMODEL * 2;
  unsigned short* w2bf = (unsigned short*)(ws + off); off += (size_t)NEXP * DMODEL * IFFN * 2;
  unsigned short* xbf  = (unsigned short*)(ws + off); off += (size_t)N_TOK * DMODEL * 2;
  float*          slot = (float*)(ws + off);          off += (size_t)N_TOK * 2 * DMODEL * 4;
  int*            cnts = (int*)(ws + off);            off += 256;
  int*            ltok = (int*)(ws + off);            off += (size_t)NEXP * N_TOK * 4;
  float*          lwt  = (float*)(ws + off);          off += (size_t)NEXP * N_TOK * 4;

  moe_zero_counts<<<1, 32, 0, stream>>>(cnts);

  const int n4w = NEXP * IFFN * DMODEL / 4;
  moe_cvt_bf16<<<8192, 256, 0, stream>>>((const float4*)w1, (uint2*)w1bf, n4w);
  moe_cvt_bf16<<<8192, 256, 0, stream>>>((const float4*)w2, (uint2*)w2bf, n4w);
  moe_cvt_bf16<<<1024, 256, 0, stream>>>((const float4*)x,  (uint2*)xbf,  N_TOK * DMODEL / 4);

  moe_gate<<<N_TOK / 256, 256, 0, stream>>>(x, gw, gb, cnts, ltok, lwt);

  moe_ffn_wmma<<<NEXP * TILES_PER_E, 256, TILE_M * IFFN * 2, stream>>>(
      w1bf, w2bf, xbf, b1, b2, cnts, ltok, lwt, slot);

  moe_combine<<<(N_TOK * DMODEL / 4) / 256, 256, 0, stream>>>((const float4*)slot, (float4*)out);
}
